// TabM_17643725652162
// MI455X (gfx1250) — matrix operand order repaired and checked by reading, not compiled, not run on hardware
//
#include <hip/hip_runtime.h>

// B=65536, F=64, NBINS=16, DEMB=8 -> DIN=512, M=32, H=32
// out: (B, M) f32, out[b*32 + m]

typedef __attribute__((ext_vector_type(8)))  _Float16 v8h;
typedef __attribute__((ext_vector_type(16))) _Float16 v16h;
typedef __attribute__((ext_vector_type(8)))  float    v8f;

#define HSTR 40    // LDS h-tile row stride in halves (bank padding)
#define XSTR 520   // LDS x-tile row stride in halves (512 + 8 pad -> 4-bank skew)

__device__ __forceinline__ v16h cat8(v8h a, v8h b) {
    return __builtin_shufflevector(a, b, 0,1,2,3,4,5,6,7,8,9,10,11,12,13,14,15);
}

__device__ __forceinline__ v8f wmma16(v16h a, v16h b, v8f c) {
    // D = A(16x32 f16) * B(32x16 f16) + C(16x16 f32)
    return __builtin_amdgcn_wmma_f32_16x16x32_f16(
        false, a, false, b, (short)0, c, false, false);
}

// B-fragment: weights laid out [.. tile ..][n(32)][k(32)] in halves.
// lane l<16 : col=l,    K=0..7,16..23 ; lane>=16: col=l-16, K=8..15,24..31
__device__ __forceinline__ v16h load_bfrag(const _Float16* wp, int col, int hi) {
    const _Float16* p = wp + col * 32 + hi * 8;
    return cat8(*(const v8h*)p, *(const v8h*)(p + 16));
}

// A-fragment (16x32 f16) from an LDS tile with row stride `stride` halves.
// lane l<16: row=l, halves = K k0..k0+7 | k0+16..k0+23 ; lane>=16: +8.
__device__ __forceinline__ v16h load_afrag_lds(const _Float16* base, int stride,
                                               int k0, int l15, int hi) {
    const _Float16* p = base + l15 * stride + k0 + hi * 8;
    return cat8(*(const v8h*)p, *(const v8h*)(p + 16));
}

// C-frag (f32 16x16): vgpr v -> row v + hi*8, col = l15.  Fused bias+relu,
// f16 store into LDS h-tile (row-major, stride HSTR).
__device__ __forceinline__ void store_relu(_Float16 (*hw)[HSTR], v8f c0, v8f c1,
                                           const float* bias, int l15, int hi) {
    float bb0 = bias[l15];
    float bb1 = bias[16 + l15];
#pragma unroll
    for (int v = 0; v < 8; ++v) {
        int r = v + hi * 8;
        hw[r][l15]      = (_Float16)fmaxf(c0[v] + bb0, 0.0f);
        hw[r][16 + l15] = (_Float16)fmaxf(c1[v] + bb1, 0.0f);
    }
}

// One 32x32 hidden layer on a 32-row h tile (two 16-row A tiles), 2x2 blocking.
__device__ __forceinline__ void hidden_layer(_Float16 (*hw)[HSTR],
                                             const _Float16* wp, const float* bias,
                                             int l15, int hi) {
    v16h ah0 = load_afrag_lds(&hw[0][0],  HSTR, 0, l15, hi);
    v16h ah1 = load_afrag_lds(&hw[16][0], HSTR, 0, l15, hi);
    v16h bf0 = load_bfrag(wp, l15, hi);
    v16h bf1 = load_bfrag(wp, 16 + l15, hi);
    v8f z = {};
    v8f c00 = wmma16(ah0, bf0, z);
    v8f c01 = wmma16(ah0, bf1, z);
    v8f c10 = wmma16(ah1, bf0, z);
    v8f c11 = wmma16(ah1, bf1, z);
    store_relu(hw,      c00, c01, bias, l15, hi);
    store_relu(hw + 16, c10, c11, bias, l15, hi);
}

// ---------------- Kernel 1: piecewise-linear embedding -> x_flat f16 ----------
__global__ void __launch_bounds__(256) emb_kernel(
    const float* __restrict__ x_cont, const float* __restrict__ edges,
    const float* __restrict__ w0, const float* __restrict__ b0,
    const float* __restrict__ wpl, _Float16* __restrict__ xf)
{
    int tid = blockIdx.x * 256 + threadIdx.x;   // B*F threads
    int f = tid & 63;
    float x = x_cont[tid];
    float acc[8];
#pragma unroll
    for (int d = 0; d < 8; ++d) acc[d] = x * w0[f * 8 + d] + b0[f * 8 + d];
    const float* eg = edges + f * 17;
    const float* wp = wpl + f * 128;   // [f][16][8]
#pragma unroll
    for (int k = 0; k < 16; ++k) {
        float left  = eg[k];
        float width = eg[k + 1] - left;
        float r = (x - left) / width;
        float e = (k == 0)  ? fminf(r, 1.0f)
                : (k == 15) ? fmaxf(r, 0.0f)
                            : fminf(fmaxf(r, 0.0f), 1.0f);
#pragma unroll
        for (int d = 0; d < 8; ++d) acc[d] += e * wp[k * 8 + d];
    }
    _Float16* o = xf + (size_t)tid * 8;
#pragma unroll
    for (int d = 0; d < 8; ++d) o[d] = (_Float16)acc[d];
}

// ---------------- Kernel 2a: W1h[m][kt(16)][n(32)][k(32)] = scaling*W1 (f16) --
__global__ void __launch_bounds__(256) prep_w1(
    const float* __restrict__ scaling, const float* __restrict__ W1,
    _Float16* __restrict__ w1h)
{
    int idx = blockIdx.x * 256 + threadIdx.x;   // 524288
    int kin = idx & 31;
    int n   = (idx >> 5) & 31;
    int kt  = (idx >> 10) & 15;
    int m   = idx >> 14;
    int k = kt * 32 + kin;
    float v = scaling[m * 512 + k] * W1[((size_t)m * 512 + k) * 32 + n];
    w1h[idx] = (_Float16)v;
}

// ---------------- Kernel 2b: W{2,3,4}h[m][n(32)][k(32)] (f16) ----------------
__global__ void __launch_bounds__(256) prep_w234(
    const float* __restrict__ W2, const float* __restrict__ W3,
    const float* __restrict__ W4, _Float16* __restrict__ wh)
{
    int idx = blockIdx.x * 256 + threadIdx.x;   // 3*32768
    int w = idx >> 15;
    int r = idx & 32767;
    int kin = r & 31;
    int n   = (r >> 5) & 31;
    int m   = r >> 10;
    const float* src = (w == 0) ? W2 : ((w == 1) ? W3 : W4);
    wh[idx] = (_Float16)src[(m * 32 + kin) * 32 + n];
}

// ---------------- Kernel 3: fused ensemble MLP, 32 rows/wave, 2x2 WMMA tiles --
__global__ void __launch_bounds__(64) mlp_kernel(
    const _Float16* __restrict__ xf,
    const _Float16* __restrict__ w1h, const _Float16* __restrict__ w2h,
    const _Float16* __restrict__ w3h, const _Float16* __restrict__ w4h,
    const float* __restrict__ b1, const float* __restrict__ b2,
    const float* __restrict__ b3, const float* __restrict__ b4,
    const float* __restrict__ w5, const float* __restrict__ b5,
    float* __restrict__ out)
{
    __shared__ __align__(16) _Float16 xlds[2][16][XSTR];  // tile-1 A rows
    __shared__ __align__(16) _Float16 hbuf[2][32][HSTR];  // 32-row h tile
    const int lane = threadIdx.x & 31;
    const int wave = threadIdx.x >> 5;
    const int l15  = lane & 15;
    const int hi   = lane >> 4;
    const int row0 = (blockIdx.x * 2 + wave) * 32;   // 32 rows per wave

    // ---- tile-0 A (rows row0..row0+15): resident in 128 VGPRs, reused 32x
    v16h a0[16];
    const _Float16* xrow = xf + (size_t)(row0 + l15) * 512;
#pragma unroll
    for (int kt = 0; kt < 16; ++kt) {
        const _Float16* p = xrow + kt * 32 + hi * 8;
        a0[kt] = cat8(*(const v8h*)p, *(const v8h*)(p + 16));
    }

    // ---- tile-1 A (rows row0+16..row0+31): copy once into LDS (row-major)
    {
        const _Float16* g = xf + (size_t)(row0 + 16 + l15) * 512 + hi * 256;
        _Float16* s = &xlds[wave][l15][hi * 256];
#pragma unroll
        for (int i = 0; i < 32; ++i)
            *(v8h*)(s + i * 8) = *(const v8h*)(g + i * 8);
    }

    const _Float16* x1 = &xlds[wave][0][0];
    _Float16 (*hw)[HSTR] = hbuf[wave];

#pragma unroll 1
    for (int m = 0; m < 32; ++m) {
        // ---- layer 1: (32x512)@(512x32); each B frag feeds 2 row tiles
        v8f c00 = {}, c01 = {}, c10 = {}, c11 = {};
        const _Float16* wp = w1h + m * (16 * 32 * 32);
#pragma unroll
        for (int kt = 0; kt < 16; ++kt) {
            v16h bf0 = load_bfrag(wp + kt * 1024, l15, hi);
            v16h bf1 = load_bfrag(wp + kt * 1024, 16 + l15, hi);
            v16h a1  = load_afrag_lds(x1, XSTR, kt * 32, l15, hi);
            c00 = wmma16(a0[kt], bf0, c00);
            c01 = wmma16(a0[kt], bf1, c01);
            c10 = wmma16(a1,     bf0, c10);
            c11 = wmma16(a1,     bf1, c11);
        }
        store_relu(hw,      c00, c01, b1 + m * 32, l15, hi);
        store_relu(hw + 16, c10, c11, b1 + m * 32, l15, hi);

        // ---- layers 2..4: 32x32 on the 32-row h tile
        hidden_layer(hw, w2h + m * 1024, b2 + m * 32, l15, hi);
        hidden_layer(hw, w3h + m * 1024, b3 + m * 32, l15, hi);
        hidden_layer(hw, w4h + m * 1024, b4 + m * 32, l15, hi);

        // ---- layer 5: H->1, every lane handles one of the 32 rows
        {
            const float* w5r = w5 + m * 32;
            float s = b5[m];
#pragma unroll
            for (int h = 0; h < 32; ++h) s += (float)hw[lane][h] * w5r[h];
            out[(size_t)(row0 + lane) * 32 + m] = s;
        }
    }
}

extern "C" void kernel_launch(void* const* d_in, const int* in_sizes, int n_in,
                              void* d_out, int out_size, void* d_ws, size_t ws_size,
                              hipStream_t stream) {
    const float* x_cont  = (const float*)d_in[0];
    // d_in[1] = x_cat (B,0) unused
    const float* edges   = (const float*)d_in[2];
    const float* w0      = (const float*)d_in[3];
    const float* b0      = (const float*)d_in[4];
    const float* wpl     = (const float*)d_in[5];
    const float* scaling = (const float*)d_in[6];
    const float* W1      = (const float*)d_in[7];
    const float* b1      = (const float*)d_in[8];
    const float* W2      = (const float*)d_in[9];
    const float* b2      = (const float*)d_in[10];
    const float* W3      = (const float*)d_in[11];
    const float* b3      = (const float*)d_in[12];
    const float* W4      = (const float*)d_in[13];
    const float* b4      = (const float*)d_in[14];
    const float* W5      = (const float*)d_in[15];
    const float* b5      = (const float*)d_in[16];
    float* out = (float*)d_out;

    char* ws = (char*)d_ws;
    _Float16* xfp  = (_Float16*)ws;                              // 64 MiB
    _Float16* w1h  = (_Float16*)(ws + 67108864);                 // 1 MiB
    _Float16* w234 = (_Float16*)(ws + 67108864 + 1048576);       // 192 KiB

    emb_kernel<<<16384, 256, 0, stream>>>(x_cont, edges, w0, b0, wpl, xfp);
    prep_w1  <<<2048,  256, 0, stream>>>(scaling, W1, w1h);
    prep_w234<<<384,   256, 0, stream>>>(W2, W3, W4, w234);
    mlp_kernel<<<1024, 64, 0, stream>>>(xfp, w1h, w234, w234 + 32768, w234 + 65536,
                                        b1, b2, b3, b4, W5, b5, out);
}
